// CausalMultiHeadSelfAttention_17265768530500
// MI455X (gfx1250) — matrix operand order repaired and checked by reading, not compiled, not run on hardware
//
#include <hip/hip_runtime.h>

// ---------------------------------------------------------------------------
// Causal multi-head self-attention, CDNA5 (gfx1250) WMMA implementation.
// B=2, S=2048, D=1024, H=16, Dh=64. fp32 in/out, f16 WMMA with f32 accum.
// ---------------------------------------------------------------------------

typedef __attribute__((ext_vector_type(16))) _Float16 v16h;
typedef __attribute__((ext_vector_type(8)))  _Float16 v8h;
typedef __attribute__((ext_vector_type(4)))  _Float16 v4h;
typedef __attribute__((ext_vector_type(8)))  float    v8f;

#define D_MODEL 1024
#define NHEADS  16
#define HDIM    64
#define SEQ     2048
#define BATCH   2
#define MROWS   (BATCH * SEQ)   // 4096

#define WMMA_F16(a, b, c) \
  __builtin_amdgcn_wmma_f32_16x16x32_f16(false, (a), false, (b), (short)0, (c), false, false)

// A-fragment (16x32, f16), source row-major with leading dim `ld` (halves).
// Lane layout (ISA 7.12.2): lanes 0-15 row M=lane, halves 0-7 = K 0..7,
// halves 8-15 = K 16..23; lanes 16-31 same rows, K 8..15 / 24..31.
__device__ __forceinline__ v16h load_a_frag(const _Float16* p, int ld) {
  const int lane = threadIdx.x & 31;
  const int r = lane & 15;
  const int hi = lane >> 4;
  const _Float16* row = p + (size_t)r * ld + hi * 8;
  v8h lo = *(const v8h*)(row);
  v8h hv = *(const v8h*)(row + 16);
  v16h a;
#pragma unroll
  for (int i = 0; i < 8; ++i) { a[i] = lo[i]; a[i + 8] = hv[i]; }
  return a;
}

// B-fragment (32x16, f16), source stored column-contiguous-in-K:
// column n at p + n*ld, K contiguous. Lanes 0-15: col=lane, K 0..7 / 16..23;
// lanes 16-31: col=lane-16, K 8..15 / 24..31.
__device__ __forceinline__ v16h load_b_frag(const _Float16* p, int ld) {
  const int lane = threadIdx.x & 31;
  const int n = lane & 15;
  const int hi = lane >> 4;
  const _Float16* col = p + (size_t)n * ld + hi * 8;
  v8h lo = *(const v8h*)(col);
  v8h hv = *(const v8h*)(col + 16);
  v16h b;
#pragma unroll
  for (int i = 0; i < 8; ++i) { b[i] = lo[i]; b[i + 8] = hv[i]; }
  return b;
}

// ---------------------------------------------------------------------------
// fp32 -> f16 cast, 4 elements/thread
// ---------------------------------------------------------------------------
__global__ void cast_f32_f16_v4(const float* __restrict__ in,
                                _Float16* __restrict__ out, int n4) {
  int i = blockIdx.x * blockDim.x + threadIdx.x;
  if (i >= n4) return;
  float4 f = ((const float4*)in)[i];
  v4h h;
  h[0] = (_Float16)f.x; h[1] = (_Float16)f.y;
  h[2] = (_Float16)f.z; h[3] = (_Float16)f.w;
  ((v4h*)out)[i] = h;
}

// ---------------------------------------------------------------------------
// QKV projection: qkv[s,o] = sum_i xh[s,i] * w[o,i]
// One 32x64 output tile per wave (2 A-frags x 4 B-frags = 8 WMMA per k-step,
// 1.5 b128 loads per WMMA). Output scattered into Q/K/V [B,H,S,Dh] (f16).
// ---------------------------------------------------------------------------
__global__ void qkv_gemm_kernel(const _Float16* __restrict__ xh,   // [4096,1024]
                                const _Float16* __restrict__ wh,   // [3072,1024]
                                _Float16* __restrict__ qg,
                                _Float16* __restrict__ kg,
                                _Float16* __restrict__ vg) {
  const int NT = (3 * D_MODEL) / 64;  // 48 column tiles of 64
  int wid = blockIdx.x * (blockDim.x >> 5) + (threadIdx.x >> 5);
  int tn = wid % NT;
  int tm = wid / NT;
  const int lane = threadIdx.x & 31;
  const int n = lane & 15, hi = lane >> 4;

  const _Float16* arow = xh + (size_t)tm * 32 * D_MODEL;
  const _Float16* bcol = wh + (size_t)tn * 64 * D_MODEL;

  v8f acc[2][4] = {};
  for (int k0 = 0; k0 < D_MODEL; k0 += 32) {
    v16h a0 = load_a_frag(arow + k0, D_MODEL);
    v16h a1 = load_a_frag(arow + 16 * D_MODEL + k0, D_MODEL);
    v16h b[4];
#pragma unroll
    for (int ni = 0; ni < 4; ++ni)
      b[ni] = load_b_frag(bcol + (size_t)ni * 16 * D_MODEL + k0, D_MODEL);
#pragma unroll
    for (int ni = 0; ni < 4; ++ni) {
      acc[0][ni] = WMMA_F16(a0, b[ni], acc[0][ni]);
      acc[1][ni] = WMMA_F16(a1, b[ni], acc[1][ni]);
    }
  }

#pragma unroll
  for (int mi = 0; mi < 2; ++mi)
#pragma unroll
    for (int ni = 0; ni < 4; ++ni) {
      int o = tn * 64 + ni * 16 + n;       // output feature 0..3071
      int sel = o >> 10;                   // 0=Q 1=K 2=V
      int oc = o & (D_MODEL - 1);
      int h = oc >> 6, d = oc & 63;
      _Float16* dst = (sel == 0) ? qg : (sel == 1) ? kg : vg;
#pragma unroll
      for (int j = 0; j < 8; ++j) {
        int row = tm * 32 + mi * 16 + hi * 8 + j;  // 0..4095
        int b_ = row >> 11;                        // batch
        int s = row & (SEQ - 1);
        dst[((size_t)(b_ * NHEADS + h) * SEQ + s) * HDIM + d] =
            (_Float16)acc[mi][ni][j];
      }
    }
}

// ---------------------------------------------------------------------------
// Flash-attention: block = 8 waves x 16 queries (128 q/block), key tiles of 64.
// ---------------------------------------------------------------------------
__global__ void attn_kernel(const _Float16* __restrict__ qg,  // [B,H,S,Dh]
                            const _Float16* __restrict__ kg,
                            const _Float16* __restrict__ vg,
                            _Float16* __restrict__ ao) {      // [B,S,D]
  __shared__ __align__(16) _Float16 lds_k[64 * 64];      // [t][d]
  __shared__ __align__(16) _Float16 lds_v[64 * 64];      // [d][t] (transposed)
  __shared__ __align__(16) _Float16 lds_p[8][16 * 64];   // per-wave P tile

  const int bh = blockIdx.x;                 // b*16 + h
  const int qblk = blockIdx.y * 128;
  const int wave = threadIdx.x >> 5;
  const int lane = threadIdx.x & 31;
  const int n = lane & 15, hi = lane >> 4;
  const int q0 = qblk + wave * 16;

  const _Float16* qbase = qg + (size_t)bh * SEQ * HDIM;
  const _Float16* kbase = kg + (size_t)bh * SEQ * HDIM;
  const _Float16* vbase = vg + (size_t)bh * SEQ * HDIM;

  // Preload Q A-fragments for this wave (Dh=64 -> two 32-wide K steps).
  v16h qa0 = load_a_frag(qbase + (size_t)q0 * HDIM + 0, HDIM);
  v16h qa1 = load_a_frag(qbase + (size_t)q0 * HDIM + 32, HDIM);

  float m[8], l[8];
  v8f oacc[4] = {};
#pragma unroll
  for (int j = 0; j < 8; ++j) { m[j] = -1e30f; l[j] = 0.0f; }

  const float scale = 0.125f;  // 1/sqrt(64)

  const int ktEnd = 2 * (int)blockIdx.y + 1;  // keys up to qblk+127
  for (int kt = 0; kt <= ktEnd; ++kt) {
    const int t0 = kt * 64;

    __syncthreads();  // previous tile fully consumed before overwrite
    // Cooperative stage of K tile ([t][d]) and transposed V tile ([d][t]).
#pragma unroll
    for (int it = 0; it < 2; ++it) {
      int idx = it * 256 + threadIdx.x;     // 0..511 covers 64 rows x 8 chunks
      int t = idx >> 3;
      int d0 = (idx & 7) << 3;
      v8h kk = *(const v8h*)(kbase + (size_t)(t0 + t) * HDIM + d0);
      *(v8h*)(&lds_k[t * 64 + d0]) = kk;
      v8h vv = *(const v8h*)(vbase + (size_t)(t0 + t) * HDIM + d0);
#pragma unroll
      for (int e = 0; e < 8; ++e) lds_v[(d0 + e) * 64 + t] = vv[e];
    }
    __syncthreads();

    // ---- logits = Q * K^T for 16 queries x 64 keys ----
    v8f cfr[4];
    float tmax[8];
#pragma unroll
    for (int j = 0; j < 8; ++j) tmax[j] = -1e30f;

#pragma unroll
    for (int nt = 0; nt < 4; ++nt) {
      v8f acc = {};
      v16h b0 = load_b_frag(lds_k + nt * 16 * 64 + 0, 64);
      v16h b1 = load_b_frag(lds_k + nt * 16 * 64 + 32, 64);
      acc = WMMA_F16(qa0, b0, acc);
      acc = WMMA_F16(qa1, b1, acc);
      int t = t0 + nt * 16 + n;
#pragma unroll
      for (int j = 0; j < 8; ++j) {
        int qr = q0 + hi * 8 + j;
        float c = acc[j] * scale;
        if (t > qr) c = -1e30f;           // causal mask
        cfr[nt][j] = c;
        tmax[j] = fmaxf(tmax[j], c);
      }
    }

    // Row-max reduce across the 16 lanes of each half-group.
#pragma unroll
    for (int j = 0; j < 8; ++j) {
#pragma unroll
      for (int s = 1; s < 16; s <<= 1)
        tmax[j] = fmaxf(tmax[j], __shfl_xor(tmax[j], s, 32));
    }

    float alpha[8], rsum[8];
#pragma unroll
    for (int j = 0; j < 8; ++j) {
      float mn = fmaxf(m[j], tmax[j]);
      alpha[j] = __expf(m[j] - mn);
      m[j] = mn;
      rsum[j] = 0.0f;
    }

#pragma unroll
    for (int nt = 0; nt < 4; ++nt)
#pragma unroll
      for (int j = 0; j < 8; ++j) {
        float p = __expf(cfr[nt][j] - m[j]);
        cfr[nt][j] = p;
        rsum[j] += p;
      }

#pragma unroll
    for (int j = 0; j < 8; ++j) {
#pragma unroll
      for (int s = 1; s < 16; s <<= 1)
        rsum[j] += __shfl_xor(rsum[j], s, 32);
      l[j] = l[j] * alpha[j] + rsum[j];
    }

#pragma unroll
    for (int dt = 0; dt < 4; ++dt)
#pragma unroll
      for (int j = 0; j < 8; ++j) oacc[dt][j] *= alpha[j];

    // ---- P (C-layout f32) -> LDS (row-major f16) -> A-fragments ----
    _Float16* pw = &lds_p[wave][0];
#pragma unroll
    for (int nt = 0; nt < 4; ++nt)
#pragma unroll
      for (int j = 0; j < 8; ++j)
        pw[(hi * 8 + j) * 64 + nt * 16 + n] = (_Float16)cfr[nt][j];
    // LDS ops from the same wave are in-order; compiler inserts s_wait_dscnt.
    v16h pa0 = load_a_frag(pw + 0, 64);
    v16h pa1 = load_a_frag(pw + 32, 64);

    // ---- O += P * V ----
#pragma unroll
    for (int dt = 0; dt < 4; ++dt) {
      v16h b0 = load_b_frag(lds_v + dt * 16 * 64 + 0, 64);
      v16h b1 = load_b_frag(lds_v + dt * 16 * 64 + 32, 64);
      oacc[dt] = WMMA_F16(pa0, b0, oacc[dt]);
      oacc[dt] = WMMA_F16(pa1, b1, oacc[dt]);
    }
  }

  // Normalize and write merged-head output [B,S,D] (f16).
  const int b_ = bh >> 4, h = bh & 15;
#pragma unroll
  for (int dt = 0; dt < 4; ++dt)
#pragma unroll
    for (int j = 0; j < 8; ++j) {
      int srow = q0 + hi * 8 + j;
      float val = oacc[dt][j] / l[j];
      ao[((size_t)(b_ * SEQ + srow)) * D_MODEL + h * HDIM + dt * 16 + n] =
          (_Float16)val;
    }
}

// ---------------------------------------------------------------------------
// Output projection: y[s,o] = sum_i ao[s,i] * w_out[o,i]  (fp32 out)
// 32x64 tile per wave, same structure as qkv_gemm.
// ---------------------------------------------------------------------------
__global__ void out_gemm_kernel(const _Float16* __restrict__ ah,  // [4096,1024]
                                const _Float16* __restrict__ wh,  // [1024,1024]
                                float* __restrict__ y) {          // [4096,1024]
  const int NT = D_MODEL / 64;  // 16 column tiles
  int wid = blockIdx.x * (blockDim.x >> 5) + (threadIdx.x >> 5);
  int tn = wid % NT;
  int tm = wid / NT;
  const int lane = threadIdx.x & 31;
  const int n = lane & 15, hi = lane >> 4;

  const _Float16* arow = ah + (size_t)tm * 32 * D_MODEL;
  const _Float16* bcol = wh + (size_t)tn * 64 * D_MODEL;

  v8f acc[2][4] = {};
  for (int k0 = 0; k0 < D_MODEL; k0 += 32) {
    v16h a0 = load_a_frag(arow + k0, D_MODEL);
    v16h a1 = load_a_frag(arow + 16 * D_MODEL + k0, D_MODEL);
    v16h b[4];
#pragma unroll
    for (int ni = 0; ni < 4; ++ni)
      b[ni] = load_b_frag(bcol + (size_t)ni * 16 * D_MODEL + k0, D_MODEL);
#pragma unroll
    for (int ni = 0; ni < 4; ++ni) {
      acc[0][ni] = WMMA_F16(a0, b[ni], acc[0][ni]);
      acc[1][ni] = WMMA_F16(a1, b[ni], acc[1][ni]);
    }
  }

#pragma unroll
  for (int mi = 0; mi < 2; ++mi)
#pragma unroll
    for (int ni = 0; ni < 4; ++ni) {
      int o = tn * 64 + ni * 16 + n;
#pragma unroll
      for (int j = 0; j < 8; ++j) {
        int row = tm * 32 + mi * 16 + hi * 8 + j;
        y[(size_t)row * D_MODEL + o] = acc[mi][ni][j];
      }
    }
}

// ---------------------------------------------------------------------------
// Launcher
// ---------------------------------------------------------------------------
extern "C" void kernel_launch(void* const* d_in, const int* in_sizes, int n_in,
                              void* d_out, int out_size, void* d_ws,
                              size_t ws_size, hipStream_t stream) {
  const float* x = (const float*)d_in[0];
  const float* wqkv = (const float*)d_in[1];
  const float* wout = (const float*)d_in[2];
  float* y = (float*)d_out;

  _Float16* base = (_Float16*)d_ws;
  size_t off = 0;
  _Float16* xh = base + off;    off += (size_t)MROWS * D_MODEL;
  _Float16* wqkvh = base + off; off += (size_t)3 * D_MODEL * D_MODEL;
  _Float16* wouth = base + off; off += (size_t)D_MODEL * D_MODEL;
  _Float16* qh = base + off;    off += (size_t)BATCH * NHEADS * SEQ * HDIM;
  _Float16* kh = base + off;    off += (size_t)BATCH * NHEADS * SEQ * HDIM;
  _Float16* vh = base + off;    off += (size_t)BATCH * NHEADS * SEQ * HDIM;
  _Float16* aoh = base + off;   off += (size_t)MROWS * D_MODEL;

  // 1) cast inputs to f16
  {
    int n4 = (MROWS * D_MODEL) / 4;
    cast_f32_f16_v4<<<(n4 + 255) / 256, 256, 0, stream>>>(x, xh, n4);
    n4 = (3 * D_MODEL * D_MODEL) / 4;
    cast_f32_f16_v4<<<(n4 + 255) / 256, 256, 0, stream>>>(wqkv, wqkvh, n4);
    n4 = (D_MODEL * D_MODEL) / 4;
    cast_f32_f16_v4<<<(n4 + 255) / 256, 256, 0, stream>>>(wout, wouth, n4);
  }

  // 2) QKV projection: (4096/32) x (3072/64) = 6144 wave-tiles, 8 waves/block
  {
    int tiles = (MROWS / 32) * ((3 * D_MODEL) / 64);
    qkv_gemm_kernel<<<tiles / 8, 256, 0, stream>>>(xh, wqkvh, qh, kh, vh);
  }

  // 3) flash attention: (B*H) x (S/128) blocks of 8 waves
  {
    dim3 grid(BATCH * NHEADS, SEQ / 128);
    attn_kernel<<<grid, 256, 0, stream>>>(qh, kh, vh, aoh);
  }

  // 4) output projection: (4096/32) x (1024/64) = 2048 wave-tiles
  {
    int tiles = (MROWS / 32) * (D_MODEL / 64);
    out_gemm_kernel<<<tiles / 8, 256, 0, stream>>>(aoh, wouth, y);
  }
}
